// GranularMoELayer_3504693314072
// MI455X (gfx1250) — compile-verified, matrix operand order checked by reading, not run on hardware
//
#include <hip/hip_runtime.h>

// Problem constants (match reference)
#define B_TOK   8192
#define D_IN    1024
#define D_OUTC  1024
#define NEXP    8
#define TOPK    2
#define H_DIM   2048

// GEMM tiling
#define BM   128
#define BN   128
#define BKK  32
#define LDST 40   // LDS row stride in bf16 elems (80 bytes: 16B-aligned rows)

typedef __attribute__((ext_vector_type(8)))  float  v8f;
typedef __attribute__((ext_vector_type(8)))  __bf16 v8bf;
typedef __attribute__((ext_vector_type(16))) __bf16 v16bf;

__device__ __forceinline__ unsigned short bf_bits(float f) {
  union { float f; unsigned u; } v; v.f = f;
  unsigned r = v.u + 0x7FFFu + ((v.u >> 16) & 1u);   // RNE
  return (unsigned short)(r >> 16);
}
__device__ __forceinline__ __bf16 f2bf(float f) {
  unsigned short s = bf_bits(f);
  return __builtin_bit_cast(__bf16, s);
}
// Pack 4 floats -> 4 bf16, single 8-byte store (p must be 8B aligned)
__device__ __forceinline__ void st4bf(__bf16* p, float a, float b, float c, float d) {
  unsigned long long w =
      (unsigned long long)bf_bits(a)
    | ((unsigned long long)bf_bits(b) << 16)
    | ((unsigned long long)bf_bits(c) << 32)
    | ((unsigned long long)bf_bits(d) << 48);
  *(unsigned long long*)p = w;
}

// ---------------- kernel 1: zero output + counters ----------------
__global__ void moe_init(float* __restrict__ out, int* __restrict__ counts,
                         int* __restrict__ cursor) {
  size_t i = (size_t)blockIdx.x * blockDim.x + threadIdx.x;
  const size_t n4 = (size_t)B_TOK * D_OUTC / 4;
  if (i < n4) ((float4*)out)[i] = make_float4(0.f, 0.f, 0.f, 0.f);
  if (blockIdx.x == 0 && threadIdx.x < NEXP) {
    counts[threadIdx.x] = 0;
    cursor[threadIdx.x] = 0;
  }
}

// ---------------- kernel 2: gating (one wave32 per token) ----------------
__global__ __launch_bounds__(256)
void moe_gate(const float* __restrict__ x, const float* __restrict__ Wg,
              const float* __restrict__ bg, int* __restrict__ counts,
              int* __restrict__ topk) {
  const int wave = threadIdx.x >> 5;
  const int lane = threadIdx.x & 31;
  const int b = blockIdx.x * 8 + wave;

  float acc[NEXP];
  #pragma unroll
  for (int e = 0; e < NEXP; ++e) acc[e] = 0.f;

  const float* xr = x + (size_t)b * D_IN;
  for (int i = lane; i < D_IN; i += 32) {
    const float xv = xr[i];
    const float* wr = Wg + (size_t)i * NEXP;
    #pragma unroll
    for (int e = 0; e < NEXP; ++e) acc[e] += xv * wr[e];
  }
  #pragma unroll
  for (int off = 16; off > 0; off >>= 1) {
    #pragma unroll
    for (int e = 0; e < NEXP; ++e) acc[e] += __shfl_xor(acc[e], off, 32);
  }
  if (lane == 0) {
    float s[NEXP];
    #pragma unroll
    for (int e = 0; e < NEXP; ++e) s[e] = acc[e] + bg[e];
    int e0 = 0;
    #pragma unroll
    for (int e = 1; e < NEXP; ++e) if (s[e] > s[e0]) e0 = e;   // first max wins
    int e1 = (e0 == 0) ? 1 : 0;
    #pragma unroll
    for (int e = 0; e < NEXP; ++e) if (e != e0 && s[e] > s[e1]) e1 = e;
    topk[b * 2 + 0] = e0;
    topk[b * 2 + 1] = e1;
    atomicAdd(&counts[e0], 1);
    atomicAdd(&counts[e1], 1);
  }
}

// ---------------- kernel 3: exclusive scan over 8 counts ----------------
__global__ void moe_scan(const int* __restrict__ counts, int* __restrict__ offsets) {
  if (threadIdx.x == 0 && blockIdx.x == 0) {
    int s = 0;
    for (int e = 0; e < NEXP; ++e) { offsets[e] = s; s += counts[e]; }
    offsets[NEXP] = s;
  }
}

// ---------------- kernel 4: build compact per-expert token lists ----------------
__global__ void moe_assign(const int* __restrict__ topk, const int* __restrict__ offsets,
                           int* __restrict__ cursor, int* __restrict__ list) {
  const int b = blockIdx.x * blockDim.x + threadIdx.x;
  if (b >= B_TOK) return;
  #pragma unroll
  for (int k = 0; k < TOPK; ++k) {
    const int e = topk[b * 2 + k];
    const int pos = atomicAdd(&cursor[e], 1);
    list[offsets[e] + pos] = b;
  }
}

// ---------------- kernel 5: x fp32 -> bf16 (once per launch) ----------------
__global__ void moe_cvt_x(const float* __restrict__ x, __bf16* __restrict__ xbf) {
  const size_t i = ((size_t)blockIdx.x * blockDim.x + threadIdx.x) * 4;
  if (i < (size_t)B_TOK * D_IN) {
    const float4 v = *(const float4*)(x + i);
    st4bf(&xbf[i], v.x, v.y, v.z, v.w);
  }
}

// ---------------- kernel 6: W[e] fp32 [R][C] -> bf16 transposed [C][R] ----------------
__global__ __launch_bounds__(256)
void moe_tcvt_w(const float* __restrict__ src, __bf16* __restrict__ dst,
                int R, int C) {
  __shared__ float tile[64][65];
  const int e = blockIdx.z;
  const float* s = src + (size_t)e * R * C;
  __bf16* d = dst + (size_t)e * R * C;
  const int c0 = blockIdx.x * 64, r0 = blockIdx.y * 64;
  const int tid = threadIdx.x;

  const int lc = (tid & 15) * 4;
  #pragma unroll
  for (int p = 0; p < 4; ++p) {
    const int lr = (tid >> 4) + p * 16;
    const float4 v = *(const float4*)(s + (size_t)(r0 + lr) * C + c0 + lc);
    tile[lr][lc + 0] = v.x; tile[lr][lc + 1] = v.y;
    tile[lr][lc + 2] = v.z; tile[lr][lc + 3] = v.w;
  }
  __syncthreads();
  const int orr = (tid & 15) * 4;
  #pragma unroll
  for (int p = 0; p < 4; ++p) {
    const int oc = (tid >> 4) + p * 16;
    st4bf(d + (size_t)(c0 + oc) * R + r0 + orr,
          tile[orr + 0][oc], tile[orr + 1][oc],
          tile[orr + 2][oc], tile[orr + 3][oc]);
  }
}

// ================= double-buffered bf16 WMMA GEMM kernels =================
// Block: 256 threads (8 waves, 4x2), tile 128x128, K-step 32.
// Both A and B feeds are pre-converted bf16; tile loads are pure b128 copies.

// ---------------- kernel 7: h = relu(x_gathered @ W1t[e]^T + b1[e]) ----------------
__global__ __launch_bounds__(256)
void moe_ffn1(const __bf16* __restrict__ xbf, const __bf16* __restrict__ w1t,
              const float* __restrict__ b1, const int* __restrict__ counts,
              const int* __restrict__ offsets, const int* __restrict__ list,
              __bf16* __restrict__ h) {
  const int e = blockIdx.x >> 6;
  const int mTile = blockIdx.x & 63;
  const int cnt = counts[e];
  const int m0 = mTile * BM;
  if (m0 >= cnt) return;
  int rows = cnt - m0; if (rows > BM) rows = BM;
  const int n0 = blockIdx.y * BN;
  const int seg = offsets[e] + m0;

  __shared__ __bf16 As[2][BM][LDST];
  __shared__ __bf16 Bs[2][BN][LDST];
  __shared__ int tokLds[BM];

  const int tid = threadIdx.x;
  if (tid < BM) tokLds[tid] = (tid < rows) ? list[seg + tid] : -1;
  __syncthreads();

  const __bf16* __restrict__ w1e = w1t + (size_t)e * H_DIM * D_IN;  // [H][D_IN]

  const int lane = tid & 31, wave = tid >> 5;
  const int wm = wave & 3, wn = wave >> 2;
  const int half = lane >> 4, l16 = lane & 15;

  // copy lanes: each thread moves 16 bf16 (32B) of A and of B per K-step
  const int cr = tid >> 1;
  const int cb = (tid & 1) * 16;
  const int tokA = tokLds[cr];
  const __bf16* aSrc = (tokA >= 0) ? (xbf + (size_t)tokA * D_IN + cb) : nullptr;
  const __bf16* bSrc = w1e + (size_t)(n0 + cr) * D_IN + cb;

  const v8f vzero = {0.f, 0.f, 0.f, 0.f, 0.f, 0.f, 0.f, 0.f};
  v8f acc[2][4];
  #pragma unroll
  for (int i = 0; i < 2; ++i)
    #pragma unroll
    for (int j = 0; j < 4; ++j) acc[i][j] = vzero;

  uint4 ra0, ra1, rb0, rb1;
  auto fetch = [&](int kb) {
    if (aSrc) {
      ra0 = *(const uint4*)(aSrc + kb);
      ra1 = *(const uint4*)(aSrc + kb + 8);
    } else {
      ra0 = make_uint4(0u, 0u, 0u, 0u);
      ra1 = make_uint4(0u, 0u, 0u, 0u);
    }
    rb0 = *(const uint4*)(bSrc + kb);
    rb1 = *(const uint4*)(bSrc + kb + 8);
    if (kb + BKK < D_IN) __builtin_prefetch(bSrc + kb + BKK, 0, 0);
  };
  auto stage = [&](int buf) {
    *(uint4*)&As[buf][cr][cb]     = ra0;
    *(uint4*)&As[buf][cr][cb + 8] = ra1;
    *(uint4*)&Bs[buf][cr][cb]     = rb0;
    *(uint4*)&Bs[buf][cr][cb + 8] = rb1;
  };

  fetch(0);
  stage(0);
  __syncthreads();

  int buf = 0;
  const int nIter = D_IN / BKK;
  for (int it = 0; it < nIter; ++it) {
    if (it + 1 < nIter) fetch((it + 1) * BKK);

    v16bf af[2], bfr[4];
    #pragma unroll
    for (int ms = 0; ms < 2; ++ms) {
      const int r = wm * 32 + ms * 16 + l16;
      const v8bf lo = *(const v8bf*)&As[buf][r][half * 8];
      const v8bf hi = *(const v8bf*)&As[buf][r][half * 8 + 16];
      af[ms] = __builtin_shufflevector(lo, hi, 0,1,2,3,4,5,6,7,8,9,10,11,12,13,14,15);
    }
    #pragma unroll
    for (int ns = 0; ns < 4; ++ns) {
      const int r = wn * 64 + ns * 16 + l16;
      const v8bf lo = *(const v8bf*)&Bs[buf][r][half * 8];
      const v8bf hi = *(const v8bf*)&Bs[buf][r][half * 8 + 16];
      bfr[ns] = __builtin_shufflevector(lo, hi, 0,1,2,3,4,5,6,7,8,9,10,11,12,13,14,15);
    }
    #pragma unroll
    for (int ms = 0; ms < 2; ++ms)
      #pragma unroll
      for (int ns = 0; ns < 4; ++ns)
        acc[ms][ns] = __builtin_amdgcn_wmma_f32_16x16x32_bf16(
            false, af[ms], false, bfr[ns], (short)0, acc[ms][ns], false, false);

    if (it + 1 < nIter) {
      stage(buf ^ 1);
      __syncthreads();
      buf ^= 1;
    }
  }

  #pragma unroll
  for (int ms = 0; ms < 2; ++ms)
    #pragma unroll
    for (int ns = 0; ns < 4; ++ns) {
      const int colg = n0 + wn * 64 + ns * 16 + l16;
      const float bias = b1[e * H_DIM + colg];
      #pragma unroll
      for (int v = 0; v < 8; ++v) {
        const int r = wm * 32 + ms * 16 + half * 8 + v;
        if (r < rows) {
          float val = acc[ms][ns][v] + bias;
          val = fmaxf(val, 0.f);
          h[(size_t)(seg + r) * H_DIM + colg] = f2bf(val);
        }
      }
    }
}

// ---------------- kernel 8: out += h_seg @ W2t[e]^T + b2[e] (2 atomic adds/elem) ----------------
__global__ __launch_bounds__(256)
void moe_ffn2(const __bf16* __restrict__ h, const __bf16* __restrict__ w2t,
              const float* __restrict__ b2, const int* __restrict__ counts,
              const int* __restrict__ offsets, const int* __restrict__ list,
              float* __restrict__ out) {
  const int e = blockIdx.x >> 6;
  const int mTile = blockIdx.x & 63;
  const int cnt = counts[e];
  const int m0 = mTile * BM;
  if (m0 >= cnt) return;
  int rows = cnt - m0; if (rows > BM) rows = BM;
  const int n0 = blockIdx.y * BN;
  const int seg = offsets[e] + m0;

  __shared__ __bf16 As[2][BM][LDST];
  __shared__ __bf16 Bs[2][BN][LDST];
  __shared__ int tokLds[BM];

  const int tid = threadIdx.x;
  if (tid < BM) tokLds[tid] = (tid < rows) ? list[seg + tid] : -1;
  __syncthreads();

  const __bf16* __restrict__ w2e = w2t + (size_t)e * D_OUTC * H_DIM;  // [D_OUT][H]

  const int lane = tid & 31, wave = tid >> 5;
  const int wm = wave & 3, wn = wave >> 2;
  const int half = lane >> 4, l16 = lane & 15;

  const int cr = tid >> 1;
  const int cb = (tid & 1) * 16;
  const __bf16* aSrc = (cr < rows) ? (h + (size_t)(seg + cr) * H_DIM + cb) : nullptr;
  const __bf16* bSrc = w2e + (size_t)(n0 + cr) * H_DIM + cb;

  const v8f vzero = {0.f, 0.f, 0.f, 0.f, 0.f, 0.f, 0.f, 0.f};
  v8f acc[2][4];
  #pragma unroll
  for (int i = 0; i < 2; ++i)
    #pragma unroll
    for (int j = 0; j < 4; ++j) acc[i][j] = vzero;

  uint4 ra0, ra1, rb0, rb1;
  auto fetch = [&](int kb) {
    if (aSrc) {
      ra0 = *(const uint4*)(aSrc + kb);
      ra1 = *(const uint4*)(aSrc + kb + 8);
    } else {
      ra0 = make_uint4(0u, 0u, 0u, 0u);
      ra1 = make_uint4(0u, 0u, 0u, 0u);
    }
    rb0 = *(const uint4*)(bSrc + kb);
    rb1 = *(const uint4*)(bSrc + kb + 8);
    if (kb + BKK < H_DIM) __builtin_prefetch(bSrc + kb + BKK, 0, 0);
  };
  auto stage = [&](int buf) {
    *(uint4*)&As[buf][cr][cb]     = ra0;
    *(uint4*)&As[buf][cr][cb + 8] = ra1;
    *(uint4*)&Bs[buf][cr][cb]     = rb0;
    *(uint4*)&Bs[buf][cr][cb + 8] = rb1;
  };

  fetch(0);
  stage(0);
  __syncthreads();

  int buf = 0;
  const int nIter = H_DIM / BKK;
  for (int it = 0; it < nIter; ++it) {
    if (it + 1 < nIter) fetch((it + 1) * BKK);

    v16bf af[2], bfr[4];
    #pragma unroll
    for (int ms = 0; ms < 2; ++ms) {
      const int r = wm * 32 + ms * 16 + l16;
      const v8bf lo = *(const v8bf*)&As[buf][r][half * 8];
      const v8bf hi = *(const v8bf*)&As[buf][r][half * 8 + 16];
      af[ms] = __builtin_shufflevector(lo, hi, 0,1,2,3,4,5,6,7,8,9,10,11,12,13,14,15);
    }
    #pragma unroll
    for (int ns = 0; ns < 4; ++ns) {
      const int r = wn * 64 + ns * 16 + l16;
      const v8bf lo = *(const v8bf*)&Bs[buf][r][half * 8];
      const v8bf hi = *(const v8bf*)&Bs[buf][r][half * 8 + 16];
      bfr[ns] = __builtin_shufflevector(lo, hi, 0,1,2,3,4,5,6,7,8,9,10,11,12,13,14,15);
    }
    #pragma unroll
    for (int ms = 0; ms < 2; ++ms)
      #pragma unroll
      for (int ns = 0; ns < 4; ++ns)
        acc[ms][ns] = __builtin_amdgcn_wmma_f32_16x16x32_bf16(
            false, af[ms], false, bfr[ns], (short)0, acc[ms][ns], false, false);

    if (it + 1 < nIter) {
      stage(buf ^ 1);
      __syncthreads();
      buf ^= 1;
    }
  }

  #pragma unroll
  for (int ms = 0; ms < 2; ++ms)
    #pragma unroll
    for (int ns = 0; ns < 4; ++ns) {
      const int colg = n0 + wn * 64 + ns * 16 + l16;
      const float bias = b2[e * D_OUTC + colg];
      #pragma unroll
      for (int v = 0; v < 8; ++v) {
        const int r = wm * 32 + ms * 16 + half * 8 + v;
        if (r < rows) {
          const int tok = tokLds[r];
          atomicAdd(&out[(size_t)tok * D_OUTC + colg], acc[ms][ns][v] + bias);
        }
      }
    }
}

// ---------------- launcher ----------------
extern "C" void kernel_launch(void* const* d_in, const int* in_sizes, int n_in,
                              void* d_out, int out_size, void* d_ws, size_t ws_size,
                              hipStream_t stream) {
  (void)in_sizes; (void)n_in; (void)out_size; (void)ws_size;
  const float* x  = (const float*)d_in[0];
  const float* Wg = (const float*)d_in[1];
  const float* bg = (const float*)d_in[2];
  const float* W1 = (const float*)d_in[3];
  const float* b1 = (const float*)d_in[4];
  const float* W2 = (const float*)d_in[5];
  const float* b2 = (const float*)d_in[6];
  float* out = (float*)d_out;

  // workspace layout (~151 MB)
  char* ws = (char*)d_ws;
  constexpr size_t OFF_COUNTS  = 0;
  constexpr size_t OFF_OFFSETS = 64;
  constexpr size_t OFF_CURSOR  = 128;
  constexpr size_t OFF_TOPK    = 256;
  constexpr size_t SZ_TOPK     = (size_t)B_TOK * TOPK * sizeof(int);
  constexpr size_t OFF_LIST    = OFF_TOPK + SZ_TOPK;
  constexpr size_t SZ_LIST     = (size_t)B_TOK * TOPK * sizeof(int);
  constexpr size_t OFF_H       = (OFF_LIST + SZ_LIST + 255) & ~(size_t)255;
  constexpr size_t SZ_H        = (size_t)B_TOK * TOPK * H_DIM * 2;          // 64 MB
  constexpr size_t OFF_XBF     = OFF_H + SZ_H;
  constexpr size_t SZ_XBF      = (size_t)B_TOK * D_IN * 2;                  // 16 MB
  constexpr size_t OFF_W1T     = OFF_XBF + SZ_XBF;
  constexpr size_t SZ_W1T      = (size_t)NEXP * D_IN * H_DIM * 2;           // 32 MB
  constexpr size_t OFF_W2T     = OFF_W1T + SZ_W1T;

  int*    counts  = (int*)(ws + OFF_COUNTS);
  int*    offsets = (int*)(ws + OFF_OFFSETS);
  int*    cursor  = (int*)(ws + OFF_CURSOR);
  int*    topk    = (int*)(ws + OFF_TOPK);
  int*    list    = (int*)(ws + OFF_LIST);
  __bf16* hbuf    = (__bf16*)(ws + OFF_H);
  __bf16* xbf     = (__bf16*)(ws + OFF_XBF);
  __bf16* w1t     = (__bf16*)(ws + OFF_W1T);   // [E][H][D_IN]
  __bf16* w2t     = (__bf16*)(ws + OFF_W2T);   // [E][D_OUT][H]

  const int n4 = B_TOK * D_OUTC / 4;
  moe_init<<<(n4 + 255) / 256, 256, 0, stream>>>(out, counts, cursor);
  moe_gate<<<B_TOK / 8, 256, 0, stream>>>(x, Wg, bg, counts, topk);
  moe_scan<<<1, 32, 0, stream>>>(counts, offsets);
  moe_assign<<<B_TOK / 256, 256, 0, stream>>>(topk, offsets, cursor, list);

  moe_cvt_x<<<(B_TOK * D_IN / 4 + 255) / 256, 256, 0, stream>>>(x, xbf);
  dim3 gt1(H_DIM / 64, D_IN / 64, NEXP);   // W1 [1024][2048] -> [2048][1024]
  moe_tcvt_w<<<gt1, 256, 0, stream>>>(W1, w1t, D_IN, H_DIM);
  dim3 gt2(D_OUTC / 64, H_DIM / 64, NEXP); // W2 [2048][1024] -> [1024][2048]
  moe_tcvt_w<<<gt2, 256, 0, stream>>>(W2, w2t, H_DIM, D_OUTC);

  dim3 g1(NEXP * (B_TOK / BM), H_DIM / BN);    // (512, 16)
  moe_ffn1<<<g1, 256, 0, stream>>>(xbf, w1t, b1, counts, offsets, list, hbuf);
  dim3 g2(NEXP * (B_TOK / BM), D_OUTC / BN);   // (512, 8)
  moe_ffn2<<<g2, 256, 0, stream>>>(hbuf, w2t, b2, counts, offsets, list, out);
}
